// MultiHeadAttention_41394894799052
// MI455X (gfx1250) — matrix operand order repaired and checked by reading, not compiled, not run on hardware
//
#include <hip/hip_runtime.h>
#include <hip/hip_bf16.h>

typedef __attribute__((ext_vector_type(16))) _Float16 v16h;
typedef __attribute__((ext_vector_type(8)))  _Float16 v8h;
typedef __attribute__((ext_vector_type(8)))  float    v8f;

#define T_SEQ 4096
#define D_MODEL 1024
#define N_HEADS 16
#define HEAD_DIM 64

__device__ __forceinline__ v8f wmma_f16(v16h a, v16h b, v8f c) {
  return __builtin_amdgcn_wmma_f32_16x16x32_f16(
      /*neg_a=*/false, a, /*neg_b=*/false, b,
      /*c_mod=*/(short)0, c, /*reuse_a=*/false, /*reuse_b=*/false);
}

// A-matrix fragment (16x32 f16), row-major src, row stride ld (halves).
// Layout: lane<16 -> M=lane, K = 0..15;
//         lane>=16 -> M=lane-16, K = 16..31.
__device__ __forceinline__ v16h load_a_frag(const _Float16* src, int row0, int k0, int ld) {
  int lane = threadIdx.x & 31;
  int m    = lane & 15;
  int kb   = (lane < 16) ? 0 : 16;
  const _Float16* p = src + (size_t)(row0 + m) * (size_t)ld + (size_t)k0;
  return *(const v16h*)(p + kb);
}

// B-matrix fragment (32x16 f16) where B[k][n] = W[n0+n][k0+k] for row-major W
// with row stride ld (halves). lane<16 -> N=lane, K=0..15; lane>=16 -> N=lane-16,
// K=16..31. One contiguous 32-byte load per lane.
__device__ __forceinline__ v16h load_b_frag(const _Float16* w, int n0, int k0, int ld) {
  int lane = threadIdx.x & 31;
  int n    = lane & 15;
  int ko   = (lane < 16) ? 0 : 16;
  const _Float16* p = w + (size_t)(n0 + n) * (size_t)ld + (size_t)(k0 + ko);
  return *(const v16h*)p;
}

__global__ __launch_bounds__(256) void cvt_f32_to_f16(const float* __restrict__ src,
                                                      _Float16* __restrict__ dst, int n) {
  int i = blockIdx.x * blockDim.x + threadIdx.x;
  if (i < n) dst[i] = (_Float16)src[i];
}

// QKV projection: qkv[t][e] = sum_k x[t][k] * Wqkv[e][k] + b[e]
// Each wave: 32(M) x 64(N) tile; N strip is head-aligned.
// Scatter: Q -> Qh[h][t][64] (scaled 1/8), K -> Kh[h][t][64], V -> Vt[h][64][t]
__global__ __launch_bounds__(128) void qkv_kernel(const _Float16* __restrict__ xh,
                                                  const _Float16* __restrict__ wh,
                                                  const float* __restrict__ bias,
                                                  _Float16* __restrict__ Qh,
                                                  _Float16* __restrict__ Kh,
                                                  _Float16* __restrict__ Vt) {
  int wid  = threadIdx.x >> 5;
  int lane = threadIdx.x & 31;
  int e0 = (blockIdx.x * 4 + wid) * 64;  // N strip (0..3071), head-aligned
  int t0 = blockIdx.y * 32;              // M tile
  v8f acc[2][4];
#pragma unroll
  for (int ms = 0; ms < 2; ++ms)
#pragma unroll
    for (int ns = 0; ns < 4; ++ns) acc[ms][ns] = {};

  for (int k0 = 0; k0 < D_MODEL; k0 += 32) {
    v16h a0 = load_a_frag(xh, t0, k0, D_MODEL);
    v16h a1 = load_a_frag(xh, t0 + 16, k0, D_MODEL);
#pragma unroll
    for (int ns = 0; ns < 4; ++ns) {
      v16h b = load_b_frag(wh, e0 + ns * 16, k0, D_MODEL);
      acc[0][ns] = wmma_f16(a0, b, acc[0][ns]);
      acc[1][ns] = wmma_f16(a1, b, acc[1][ns]);
    }
  }

  int n = lane & 15;
  int rbase = (lane >> 4) << 3;
  int sec  = e0 >> 10;           // uniform per wave: 0=Q 1=K 2=V
  int head = (e0 & 1023) >> 6;   // uniform per wave (strip == one head)
#pragma unroll
  for (int ms = 0; ms < 2; ++ms) {
#pragma unroll
    for (int ns = 0; ns < 4; ++ns) {
      int dcol = ns * 16 + n;
      float bv = bias[e0 + dcol];
#pragma unroll
      for (int r = 0; r < 8; ++r) {
        int t = t0 + ms * 16 + rbase + r;
        float v = acc[ms][ns][r] + bv;
        if (sec == 0) {
          Qh[((size_t)(head << 12) + t) * 64 + dcol] = (_Float16)(v * 0.125f);
        } else if (sec == 1) {
          Kh[((size_t)(head << 12) + t) * 64 + dcol] = (_Float16)v;
        } else {
          Vt[(((size_t)(head << 6) + dcol) << 12) + t] = (_Float16)v;
        }
      }
    }
  }
}

// Flash attention: each wave owns 32 query rows (two 16-row subtiles) of one
// head; K/V B-fragments are loaded once per 32-key chunk and shared by both.
__global__ __launch_bounds__(128) void attn_kernel(const _Float16* __restrict__ Qh,
                                                   const _Float16* __restrict__ Kh,
                                                   const _Float16* __restrict__ Vt,
                                                   const float* __restrict__ bias,
                                                   _Float16* __restrict__ Oh) {
  __shared__ __align__(32) _Float16 tile[4][2][16][32];
  int wid  = threadIdx.x >> 5;
  int lane = threadIdx.x & 31;
  int h  = blockIdx.y;
  int q0 = (blockIdx.x * 4 + wid) * 32;

  const _Float16* Qb = Qh + (size_t)h * T_SEQ * HEAD_DIM;
  const _Float16* Kb = Kh + (size_t)h * T_SEQ * HEAD_DIM;
  const _Float16* Vb = Vt + (size_t)h * HEAD_DIM * T_SEQ;
  const float*    bh = bias + (size_t)h * T_SEQ;

  v16h aq[2][2];
#pragma unroll
  for (int s = 0; s < 2; ++s) {
    aq[s][0] = load_a_frag(Qb, q0 + 16 * s, 0, HEAD_DIM);
    aq[s][1] = load_a_frag(Qb, q0 + 16 * s, 32, HEAD_DIM);
  }

  float mrow[2][8], lrow[2][8];
  v8f o[2][4];
#pragma unroll
  for (int s = 0; s < 2; ++s) {
#pragma unroll
    for (int r = 0; r < 8; ++r) { mrow[s][r] = -INFINITY; lrow[s][r] = 0.0f; }
#pragma unroll
    for (int d = 0; d < 4; ++d) o[s][d] = {};
  }

  int n     = lane & 15;
  int rbase = (lane >> 4) << 3;

  for (int j = 0; j <= q0 + 31; j += 32) {   // uniform per wave (causal bound)
    v16h bk00 = load_b_frag(Kb, j,      0,  HEAD_DIM);
    v16h bk01 = load_b_frag(Kb, j,      32, HEAD_DIM);
    v16h bk10 = load_b_frag(Kb, j + 16, 0,  HEAD_DIM);
    v16h bk11 = load_b_frag(Kb, j + 16, 32, HEAD_DIM);
    int c0 = j + n, c1 = j + 16 + n;
    float b0 = bh[c0], b1 = bh[c1];

#pragma unroll
    for (int s = 0; s < 2; ++s) {
      v8f s0 = {}, s1 = {};
      s0 = wmma_f16(aq[s][0], bk00, s0);
      s0 = wmma_f16(aq[s][1], bk01, s0);
      s1 = wmma_f16(aq[s][0], bk10, s1);
      s1 = wmma_f16(aq[s][1], bk11, s1);
#pragma unroll
      for (int r = 0; r < 8; ++r) {
        int row = q0 + 16 * s + rbase + r;   // j <= q0 <= row: never fully masked
        float v0 = (c0 <= row) ? (s0[r] + b0) : -3.0e38f;
        float v1 = (c1 <= row) ? (s1[r] + b1) : -3.0e38f;
        float mx = fmaxf(v0, v1);
        mx = fmaxf(mx, __shfl_xor(mx, 1));
        mx = fmaxf(mx, __shfl_xor(mx, 2));
        mx = fmaxf(mx, __shfl_xor(mx, 4));
        mx = fmaxf(mx, __shfl_xor(mx, 8));
        float mnew  = fmaxf(mrow[s][r], mx);
        float alpha = __expf(mrow[s][r] - mnew);
        float p0 = __expf(v0 - mnew);
        float p1 = __expf(v1 - mnew);
        float sum = p0 + p1;
        sum += __shfl_xor(sum, 1);
        sum += __shfl_xor(sum, 2);
        sum += __shfl_xor(sum, 4);
        sum += __shfl_xor(sum, 8);
        lrow[s][r] = lrow[s][r] * alpha + sum;
        mrow[s][r] = mnew;
        o[s][0][r] *= alpha; o[s][1][r] *= alpha;
        o[s][2][r] *= alpha; o[s][3][r] *= alpha;
        int m = rbase + r;
        tile[wid][s][m][n]      = (_Float16)p0;
        tile[wid][s][m][16 + n] = (_Float16)p1;
      }
    }
    // wave-local LDS RAW: DS ops are in-order per wave; keep compiler honest.
    asm volatile("s_wait_dscnt 0" ::: "memory");
    v16h ap0 = load_a_frag(&tile[wid][0][0][0], 0, 0, 32);
    v16h ap1 = load_a_frag(&tile[wid][1][0][0], 0, 0, 32);
#pragma unroll
    for (int d = 0; d < 4; ++d) {
      v16h bv = load_b_frag(Vb, d * 16, j, T_SEQ);
      o[0][d] = wmma_f16(ap0, bv, o[0][d]);
      o[1][d] = wmma_f16(ap1, bv, o[1][d]);
    }
  }

#pragma unroll
  for (int s = 0; s < 2; ++s) {
#pragma unroll
    for (int r = 0; r < 8; ++r) {
      float inv = 1.0f / lrow[s][r];
      int t = q0 + 16 * s + rbase + r;
      size_t base = (size_t)t * D_MODEL + (size_t)h * HEAD_DIM + n;
      Oh[base +  0] = (_Float16)(o[s][0][r] * inv);
      Oh[base + 16] = (_Float16)(o[s][1][r] * inv);
      Oh[base + 32] = (_Float16)(o[s][2][r] * inv);
      Oh[base + 48] = (_Float16)(o[s][3][r] * inv);
    }
  }
}

// Output projection: out[t][e] = sum_k O[t][k] * out_w[e][k] + out_b[e]  (f32 out)
// Each wave: 32(M) x 64(N) tile.
__global__ __launch_bounds__(128) void out_kernel(const _Float16* __restrict__ Oh,
                                                  const _Float16* __restrict__ wh,
                                                  const float* __restrict__ bias,
                                                  float* __restrict__ out) {
  int wid  = threadIdx.x >> 5;
  int lane = threadIdx.x & 31;
  int e0 = (blockIdx.x * 4 + wid) * 64;
  int t0 = blockIdx.y * 32;
  v8f acc[2][4];
#pragma unroll
  for (int ms = 0; ms < 2; ++ms)
#pragma unroll
    for (int ns = 0; ns < 4; ++ns) acc[ms][ns] = {};

  for (int k0 = 0; k0 < D_MODEL; k0 += 32) {
    v16h a0 = load_a_frag(Oh, t0, k0, D_MODEL);
    v16h a1 = load_a_frag(Oh, t0 + 16, k0, D_MODEL);
#pragma unroll
    for (int ns = 0; ns < 4; ++ns) {
      v16h b = load_b_frag(wh, e0 + ns * 16, k0, D_MODEL);
      acc[0][ns] = wmma_f16(a0, b, acc[0][ns]);
      acc[1][ns] = wmma_f16(a1, b, acc[1][ns]);
    }
  }

  int n = lane & 15;
  int rbase = (lane >> 4) << 3;
#pragma unroll
  for (int ms = 0; ms < 2; ++ms) {
#pragma unroll
    for (int ns = 0; ns < 4; ++ns) {
      float bv = bias[e0 + ns * 16 + n];
#pragma unroll
      for (int r = 0; r < 8; ++r) {
        int t = t0 + ms * 16 + rbase + r;
        out[(size_t)t * D_MODEL + e0 + ns * 16 + n] = acc[ms][ns][r] + bv;
      }
    }
  }
}

extern "C" void kernel_launch(void* const* d_in, const int* in_sizes, int n_in,
                              void* d_out, int out_size, void* d_ws, size_t ws_size,
                              hipStream_t stream) {
  const float* x         = (const float*)d_in[0];  // [1,4096,1024]
  const float* attn_bias = (const float*)d_in[1];  // [1,16,1,4096]
  const float* Wqkv_w    = (const float*)d_in[2];  // [3072,1024]
  const float* Wqkv_b    = (const float*)d_in[3];  // [3072]
  const float* out_w     = (const float*)d_in[4];  // [1024,1024]
  const float* out_b     = (const float*)d_in[5];  // [1024]
  float* out = (float*)d_out;

  char* ws = (char*)d_ws;
  _Float16* xh    = (_Float16*)(ws + 0);          // 8 MB  x as f16
  _Float16* wqkvh = (_Float16*)(ws + 8388608);    // 6 MB  Wqkv as f16
  _Float16* owh   = (_Float16*)(ws + 14680064);   // 2 MB  out_w as f16
  _Float16* Qh    = (_Float16*)(ws + 16777216);   // 8 MB  [h][t][64] (pre-scaled)
  _Float16* Kh    = (_Float16*)(ws + 25165824);   // 8 MB  [h][t][64]
  _Float16* Vt    = (_Float16*)(ws + 33554432);   // 8 MB  [h][64][t]
  _Float16* Oh    = (_Float16*)(ws + 41943040);   // 8 MB  attn out [t][1024]

  const int nX = T_SEQ * D_MODEL;        // 4194304
  const int nW = 3 * D_MODEL * D_MODEL;  // 3145728
  const int nO = D_MODEL * D_MODEL;      // 1048576
  cvt_f32_to_f16<<<(nX + 255) / 256, 256, 0, stream>>>(x, xh, nX);
  cvt_f32_to_f16<<<(nW + 255) / 256, 256, 0, stream>>>(Wqkv_w, wqkvh, nW);
  cvt_f32_to_f16<<<(nO + 255) / 256, 256, 0, stream>>>(out_w, owh, nO);

  // QKV GEMM: 48 N-strips/4 waves = 12 (grid.x), 4096/32 = 128 M tiles (grid.y)
  qkv_kernel<<<dim3(12, 128), 128, 0, stream>>>(xh, wqkvh, Wqkv_b, Qh, Kh, Vt);

  // Attention: 128 q-tiles(32 rows)/4 waves = 32 (grid.x), 16 heads (grid.y)
  attn_kernel<<<dim3(32, 16), 128, 0, stream>>>(Qh, Kh, Vt, attn_bias, Oh);

  // Out GEMM: 16 N-strips/4 waves = 4 (grid.x), 128 M tiles (grid.y)
  out_kernel<<<dim3(4, 128), 128, 0, stream>>>(Oh, owh, out_b, out);
}
